// ParameterHashingCompositionModule_38104949850564
// MI455X (gfx1250) — matrix operand order repaired and checked by reading, not compiled, not run on hardware
//
#include <hip/hip_runtime.h>
#include <hip/hip_bf16.h>

typedef _Float16 v16h __attribute__((ext_vector_type(16)));
typedef _Float16 v8h  __attribute__((ext_vector_type(8)));
typedef float    v8f  __attribute__((ext_vector_type(8)));

static constexpr int   B_   = 512;
static constexpr int   IN_T = 4096;
static constexpr int   IN_I = 2048;
static constexpr float EPS  = 1e-5f;

// ---------------------------------------------------------------------------
// Prep kernels: f32 -> f16 convert, and the hashed-weight gather
// (memory bound; ~100 MB of the ~130 MB compulsory HBM traffic lives here;
//  the 33.5 MB h_weight table stays resident in the 192 MB L2).
// ---------------------------------------------------------------------------
__global__ void f32_to_f16_kernel(const float* __restrict__ src,
                                  _Float16* __restrict__ dst, size_t n) {
  size_t i = (size_t)blockIdx.x * blockDim.x + threadIdx.x;
  size_t stride = (size_t)gridDim.x * blockDim.x;
  for (; i < n; i += stride) dst[i] = (_Float16)src[i];
}

__global__ void gather_w16_kernel(const float* __restrict__ hw,
                                  const int* __restrict__ idxW,
                                  const float* __restrict__ xiW,
                                  _Float16* __restrict__ dst, size_t n) {
  size_t i = (size_t)blockIdx.x * blockDim.x + threadIdx.x;
  size_t stride = (size_t)gridDim.x * blockDim.x;
  for (; i < n; i += stride) dst[i] = (_Float16)(hw[idxW[i]] * xiW[i]);
}

// LDS offset of a shared-memory pointer: generic AS3 addresses carry the
// aperture in the high 32 bits (ISA 10.2 aperture rules); addr[31:0] is the
// workgroup-relative LDS offset that GLOBAL_LOAD_ASYNC_TO_LDS consumes.
__device__ __forceinline__ unsigned lds_off(const void* p) {
  return (unsigned)(unsigned long long)p;
}

// GVS-form async L2->LDS copy, 16 bytes/lane, tracked by ASYNCcnt.
// s_wait_dscnt first: DS and ASYNC queues are mutually unordered, so make
// sure prior ds_load fragment reads of the buffer being overwritten retired.
__device__ __forceinline__ void async_ld_b128(unsigned dst_lds, unsigned goff,
                                              unsigned long long base) {
  asm volatile("s_wait_dscnt 0x0\n\t"
               "global_load_async_to_lds_b128 %0, %1, %2"
               :: "v"(dst_lds), "v"(goff), "s"(base) : "memory");
}
__device__ __forceinline__ void wait_async0() {
  asm volatile("s_wait_asynccnt 0x0" ::: "memory");
}

// ---------------------------------------------------------------------------
// WMMA GEMM:  C[M,N] = A[M,K] (row-major f16)  x  Bm[N,K]^T (row-major f16)
// BM=64 x BN=128 x BK=32 tile, 256 threads = 8 wave32 waves (2 M x 4 N),
// each wave: 32x32 = 2x2 v_wmma_f32_16x16x32_f16 accumulators.
// Double-buffered LDS fed by GLOBAL_LOAD_ASYNC_TO_LDS_B128.
// EPILOGUE==1: + hash-bias + f_img, BN(running stats), ReLU -> f16 y.
// EPILOGUE==2: + b2 -> f32 out.
// ---------------------------------------------------------------------------
template <int EPILOGUE>
__global__ __launch_bounds__(256)
void wmma_gemm_kernel(const _Float16* __restrict__ A,
                      const _Float16* __restrict__ Bm,
                      int M, int N, int K,
                      const float* __restrict__ f_img,
                      const float* __restrict__ h_bias,
                      const int* __restrict__ idxB,
                      const float* __restrict__ xiB,
                      const float* __restrict__ gamma,
                      const float* __restrict__ beta,
                      const float* __restrict__ rmean,
                      const float* __restrict__ rvar,
                      _Float16* __restrict__ Y16,
                      const float* __restrict__ b2,
                      float* __restrict__ out) {
  constexpr int BM = 64, BN = 128, BK = 32;
  constexpr int LDA = BK + 8;  // 80B row stride -> 20-bank step, conflict-free
  constexpr int LDB = BK + 8;  // ds_load_b128 fragment reads

  __shared__ _Float16 As[2][BM][LDA];
  __shared__ _Float16 Bs[2][BN][LDB];

  const int tid   = threadIdx.x;
  const int lane  = tid & 31;
  const int wave  = tid >> 5;
  const int wm    = wave & 1;    // 2 waves along M (32 rows each)
  const int wn    = wave >> 1;   // 4 waves along N (32 cols each)
  const int lmod  = lane & 15;
  const int lhalf = lane >> 4;   // 0: lanes 0-15, 1: lanes 16-31

  const int bm = blockIdx.x * BM;
  const int bn = blockIdx.y * BN;

  // cooperative tile-load mapping (16B async chunks per thread)
  const int a_row = tid >> 2;          // 64 rows,  4 chunks of 8 halfs
  const int a_col = (tid & 3) * 8;
  const int b_row = tid >> 1;          // 128 rows, 2x 16B chunks of 16 halfs
  const int b_col = (tid & 1) * 16;

  const unsigned long long aBase = (unsigned long long)A;
  const unsigned long long bBase = (unsigned long long)Bm;
  const unsigned aLds0 = lds_off(&As[0][a_row][a_col]);
  const unsigned bLds0 = lds_off(&Bs[0][b_row][b_col]);
  constexpr unsigned ASZ = BM * LDA * 2;   // bytes per A buffer
  constexpr unsigned BSZ = BN * LDB * 2;   // bytes per B buffer

  auto issue_tile = [&](int kb, int buf) {
    const unsigned aG = (unsigned)((((size_t)(bm + a_row)) * K + kb + a_col) * 2);
    const unsigned bG = (unsigned)((((size_t)(bn + b_row)) * K + kb + b_col) * 2);
    async_ld_b128(aLds0 + buf * ASZ, aG, aBase);
    async_ld_b128(bLds0 + buf * BSZ, bG, bBase);
    async_ld_b128(bLds0 + buf * BSZ + 16, bG + 16, bBase);
  };

  v8f acc[2][2] = {};
  const int nt = K / BK;

  issue_tile(0, 0);
  wait_async0();
  __syncthreads();

  for (int t = 0; t < nt; ++t) {
    const int cur = t & 1;
    // Prefetch next tile into the other buffer. Safe: the barrier closing
    // iteration t-1 proved every wave finished reading buffer cur^1.
    if (t + 1 < nt) issue_tile((t + 1) * BK, cur ^ 1);

    // ---- B fragments (32x16 KxN): lanes 0-15 K=0..15, lanes 16-31 K=16..31
    union { v16h v; v8h h[2]; } bf[2];
#pragma unroll
    for (int ni = 0; ni < 2; ++ni) {
      const int o = wn * 32 + ni * 16 + lmod;
      bf[ni].h[0] = *(const v8h*)&Bs[cur][o][lhalf * 16];
      bf[ni].h[1] = *(const v8h*)&Bs[cur][o][lhalf * 16 + 8];
    }
    // ---- A fragments (16x32 MxK): lanes 0-15 K=0..15, lanes 16-31 K=16..31
#pragma unroll
    for (int mi = 0; mi < 2; ++mi) {
      const int m = wm * 32 + mi * 16 + lmod;
      union { v16h v; v8h h[2]; } af;
      af.h[0] = *(const v8h*)&As[cur][m][lhalf * 16];
      af.h[1] = *(const v8h*)&As[cur][m][lhalf * 16 + 8];
#pragma unroll
      for (int ni = 0; ni < 2; ++ni) {
        acc[mi][ni] = __builtin_amdgcn_wmma_f32_16x16x32_f16(
            false, af.v, false, bf[ni].v, (short)0, acc[mi][ni], false, false);
      }
    }

    if (t + 1 < nt) {
      wait_async0();      // own async chunks of tile t+1 landed in LDS
      __syncthreads();    // everyone's landed; everyone done reading cur
    }
  }

  // ---- epilogue; C/D layout: VGPR r -> M = r + 8*(lane>=16), N = lane&15 ----
#pragma unroll
  for (int mi = 0; mi < 2; ++mi) {
#pragma unroll
    for (int ni = 0; ni < 2; ++ni) {
      const int gm0 = bm + wm * 32 + mi * 16 + lhalf * 8;
      const int go  = bn + wn * 32 + ni * 16 + lmod;
      if (EPILOGUE == 1) {
        const float hb  = h_bias[idxB[go]] * xiB[go];
        const float gmu = rmean[go];
        const float gsc = gamma[go] * __frsqrt_rn(rvar[go] + EPS);
        const float gbe = beta[go];
#pragma unroll
        for (int r = 0; r < 8; ++r) {
          const int gb = gm0 + r;
          float v = acc[mi][ni][r] + hb + f_img[(size_t)gb * IN_I + go];
          v = (v - gmu) * gsc + gbe;
          v = v > 0.0f ? v : 0.0f;
          Y16[(size_t)gb * N + go] = (_Float16)v;
        }
      } else {
        const float bias = b2[go];
#pragma unroll
        for (int r = 0; r < 8; ++r) {
          const int gb = gm0 + r;
          out[(size_t)gb * N + go] = acc[mi][ni][r] + bias;
        }
      }
    }
  }
}

// ---------------------------------------------------------------------------
// Host launcher
// ---------------------------------------------------------------------------
extern "C" void kernel_launch(void* const* d_in, const int* in_sizes, int n_in,
                              void* d_out, int out_size, void* d_ws, size_t ws_size,
                              hipStream_t stream) {
  (void)in_sizes; (void)n_in; (void)out_size; (void)ws_size;

  const float* f_img    = (const float*)d_in[0];
  const float* f_text   = (const float*)d_in[1];
  const float* h_weight = (const float*)d_in[2];
  const float* h_bias   = (const float*)d_in[3];
  const float* xiW      = (const float*)d_in[4];
  const float* xiB      = (const float*)d_in[5];
  const float* gamma    = (const float*)d_in[6];
  const float* beta     = (const float*)d_in[7];
  const float* rmean    = (const float*)d_in[8];
  const float* rvar     = (const float*)d_in[9];
  const float* W2       = (const float*)d_in[10];
  const float* b2       = (const float*)d_in[11];
  const int*   idxW     = (const int*)d_in[12];
  const int*   idxB     = (const int*)d_in[13];
  float*       out      = (float*)d_out;

  // workspace layout (f16): A16 | W16 | W2_16 | Y16  (~30 MiB total)
  _Float16* A16   = (_Float16*)d_ws;
  _Float16* W16   = A16   + (size_t)B_   * IN_T;
  _Float16* W2_16 = W16   + (size_t)IN_I * IN_T;
  _Float16* Y16   = W2_16 + (size_t)IN_I * IN_I;

  // 1) converts + hashed-weight gather
  f32_to_f16_kernel<<<512, 256, 0, stream>>>(f_text, A16, (size_t)B_ * IN_T);
  f32_to_f16_kernel<<<1024, 256, 0, stream>>>(W2, W2_16, (size_t)IN_I * IN_I);
  gather_w16_kernel<<<2048, 256, 0, stream>>>(h_weight, idxW, xiW, W16,
                                              (size_t)IN_I * IN_T);

  // 2) GEMM1 + hash-bias + f_img + BN + ReLU -> Y16
  dim3 grid1(B_ / 64, IN_I / 128);
  wmma_gemm_kernel<1><<<grid1, 256, 0, stream>>>(
      A16, W16, B_, IN_I, IN_T,
      f_img, h_bias, idxB, xiB, gamma, beta, rmean, rvar, Y16,
      nullptr, nullptr);

  // 3) GEMM2 + b2 -> out (f32)
  dim3 grid2(B_ / 64, IN_I / 128);
  wmma_gemm_kernel<2><<<grid2, 256, 0, stream>>>(
      Y16, W2_16, B_, IN_I, IN_I,
      nullptr, nullptr, nullptr, nullptr, nullptr, nullptr, nullptr, nullptr,
      nullptr, b2, out);
}
